// Model_1211180778036
// MI455X (gfx1250) — matrix operand order repaired and checked by reading, not compiled, not run on hardware
//
#include <hip/hip_runtime.h>

// ---------------------------------------------------------------------------
// Types / helpers for CDNA5 WMMA (gfx1250, wave32)
// ---------------------------------------------------------------------------
typedef __attribute__((ext_vector_type(16))) __bf16 v16bf;
typedef __attribute__((ext_vector_type(8)))  float  v8f;
typedef __attribute__((ext_vector_type(2)))  unsigned int u32x2;

enum { GF_BIAS = 1, GF_RELU = 2, GF_RESID = 4, GF_TRANSB = 8 };

__device__ __forceinline__ unsigned short f2bf(float f) {
  unsigned int u = __builtin_bit_cast(unsigned int, f);
  u += 0x7FFFu + ((u >> 16) & 1u);          // round-to-nearest-even
  return (unsigned short)(u >> 16);
}
__device__ __forceinline__ unsigned int pack_bf16(float lo, float hi) {
  return (unsigned int)f2bf(lo) | ((unsigned int)f2bf(hi) << 16);
}

union Frag16 { v16bf v; unsigned int u[8]; };

// ---------------------------------------------------------------------------
// Generic GEMM:  C[M,N] = A[M,K] * op(B)  (+bias) (+resid) (relu)
// One wave per block; wave computes a 16x64 strip of C via 4 WMMA tiles.
// fp32 operands are converted to bf16 while staging into LDS.
// Fast path: vectorized float4 loads + packed b32/b64 LDS stores for interior
// tiles; bounds-checked scalar path only for ragged edges (small GCN dims,
// last N block).
// ---------------------------------------------------------------------------
__global__ __launch_bounds__(32)
void gemm_bf16_wmma(const float* __restrict__ A, const float* __restrict__ B,
                    const float* __restrict__ bias, const float* __restrict__ resid,
                    float* __restrict__ C,
                    int M, int N, int K, int lda, int ldb, int ldc, int flags)
{
  __shared__ __align__(16) unsigned short lA[16][36];   // [m][k], row = 18 dwords
  __shared__ __align__(16) unsigned short lB[64][36];   // [n][k], row = 18 dwords

  const int lane = threadIdx.x;
  const int l16  = lane & 15;
  const int half = lane >> 4;
  const int m0   = blockIdx.y << 4;
  const int n0   = blockIdx.x << 6;
  const bool trb = (flags & GF_TRANSB) != 0;

  const bool alnA = ((lda & 3) == 0) && ((((unsigned long long)(size_t)A) & 15) == 0);
  const bool alnB = ((ldb & 3) == 0) && ((((unsigned long long)(size_t)B) & 15) == 0);

  unsigned int* dA = (unsigned int*)&lA[0][0];
  unsigned int* dB = (unsigned int*)&lB[0][0];

  v8f acc[4];
#pragma unroll
  for (int t = 0; t < 4; ++t)
#pragma unroll
    for (int r = 0; r < 8; ++r) acc[t][r] = 0.0f;

  const int ksteps = (K + 31) >> 5;
  for (int ks = 0; ks < ksteps; ++ks) {
    const int k0 = ks << 5;
    const bool fullK = (k0 + 32) <= K;
    const bool fullA = fullK && ((m0 + 16) <= M) && alnA;
    const bool fullB = fullK && ((n0 + 64) <= N) && alnB;

    // ---------------- stage A tile (16 x 32) ----------------
    if (fullA) {
#pragma unroll
      for (int it = 0; it < 4; ++it) {
        int p = lane + (it << 5);
        int m = p >> 3, kk = (p & 7) << 2;
        const float4 v = *(const float4*)(A + (size_t)(m0 + m) * lda + (k0 + kk));
        u32x2 w; w.x = pack_bf16(v.x, v.y); w.y = pack_bf16(v.z, v.w);
        *(u32x2*)&dA[m * 18 + (kk >> 1)] = w;
      }
    } else {
      for (int it = 0; it < 16; ++it) {
        int e = lane + (it << 5);
        int m = e >> 5, kk = e & 31;
        float v = 0.0f;
        if ((m0 + m) < M && (k0 + kk) < K) v = A[(size_t)(m0 + m) * lda + (k0 + kk)];
        lA[m][kk] = f2bf(v);
      }
    }

    // ---------------- stage B tile (32 x 64) into lB[n][k] ----------------
    if (trb) {                           // B given as N x K row-major (e.g. K^T)
      if (fullB) {
#pragma unroll
        for (int it = 0; it < 16; ++it) {
          int p = lane + (it << 5);
          int n = p >> 3, kk = (p & 7) << 2;
          const float4 v = *(const float4*)(B + (size_t)(n0 + n) * ldb + (k0 + kk));
          u32x2 w; w.x = pack_bf16(v.x, v.y); w.y = pack_bf16(v.z, v.w);
          *(u32x2*)&dB[n * 18 + (kk >> 1)] = w;
        }
      } else {
        for (int it = 0; it < 64; ++it) {
          int e = lane + (it << 5);
          int kk = e & 31, n = e >> 5;
          float v = 0.0f;
          if ((n0 + n) < N && (k0 + kk) < K) v = B[(size_t)(n0 + n) * ldb + (k0 + kk)];
          lB[n][kk] = f2bf(v);
        }
      }
    } else {                             // B given as K x N row-major
      if (fullB) {
#pragma unroll
        for (int it = 0; it < 8; ++it) {
          int p  = lane + (it << 5);
          int kp = p >> 4;               // K pair index 0..15
          int n4 = (p & 15) << 2;        // 0,4,...,60
          const float* r0 = B + (size_t)(k0 + 2 * kp) * ldb + (n0 + n4);
          const float4 a0 = *(const float4*)(r0);
          const float4 a1 = *(const float4*)(r0 + ldb);
          dB[(n4 + 0) * 18 + kp] = pack_bf16(a0.x, a1.x);
          dB[(n4 + 1) * 18 + kp] = pack_bf16(a0.y, a1.y);
          dB[(n4 + 2) * 18 + kp] = pack_bf16(a0.z, a1.z);
          dB[(n4 + 3) * 18 + kp] = pack_bf16(a0.w, a1.w);
        }
      } else {
        for (int it = 0; it < 64; ++it) {
          int e = lane + (it << 5);
          int n = e & 63, kk = e >> 6;
          float v = 0.0f;
          if ((k0 + kk) < K && (n0 + n) < N) v = B[(size_t)(k0 + kk) * ldb + (n0 + n)];
          lB[n][kk] = f2bf(v);
        }
      }
    }

    if (ks + 1 < ksteps) {               // pull next K-tiles toward the caches
      __builtin_prefetch(A + (size_t)(m0 + l16) * lda + (k0 + 32), 0, 0);
      if (trb) __builtin_prefetch(B + (size_t)(n0 + (lane & 31)) * ldb + (k0 + 32), 0, 0);
      else     __builtin_prefetch(B + (size_t)(k0 + 32 + half * 16) * ldb + (n0 + l16 * 4), 0, 0);
    }

    __syncthreads();

    // ---- gather A fragment: lane l16 = row M; K halves split by lane>>4 ----
    Frag16 fa;
    const unsigned int* rowA = (const unsigned int*)(&lA[l16][0]);
#pragma unroll
    for (int i = 0; i < 4; ++i) fa.u[i]     = rowA[(half << 2) + i];      // K 0..7 / 8..15
#pragma unroll
    for (int i = 0; i < 4; ++i) fa.u[4 + i] = rowA[8 + (half << 2) + i];  // K 16..23 / 24..31

    // ---- 4 N-tiles reuse the A fragment ----
#pragma unroll
    for (int t = 0; t < 4; ++t) {
      Frag16 fb;
      const unsigned int* rowB = (const unsigned int*)(&lB[(t << 4) + l16][0]);
#pragma unroll
      for (int j = 0; j < 8; ++j) fb.u[j] = rowB[(half << 2) + j + ((j >> 2) << 2)];  // K 0..7,16..23 / 8..15,24..31
      acc[t] = __builtin_amdgcn_wmma_f32_16x16x32_bf16(
                 false, fa.v, false, fb.v, (short)0, acc[t], false, false);
    }
    __syncthreads();
  }

  // ---- epilogue: C layout VGPR r -> M = r + 8*half, N = lane&15 ----
#pragma unroll
  for (int t = 0; t < 4; ++t) {
    int n = n0 + (t << 4) + l16;
    if (n >= N) continue;
#pragma unroll
    for (int r = 0; r < 8; ++r) {
      int m = m0 + r + (half << 3);
      if (m >= M) continue;
      float v = acc[t][r];
      if (flags & GF_BIAS)  v += bias[n];
      if (flags & GF_RESID) v += resid[(size_t)m * ldc + n];
      if (flags & GF_RELU)  v = fmaxf(v, 0.0f);
      C[(size_t)m * ldc + n] = v;
    }
  }
}

// ---------------------------------------------------------------------------
// Elementwise / graph kernels
// ---------------------------------------------------------------------------
__global__ void k_transpose_in(const float* __restrict__ in, float* __restrict__ out,
                               int Bn, int F, int Nn) {
  int t = blockIdx.x * blockDim.x + threadIdx.x;
  int total = Bn * F * Nn;
  if (t >= total) return;
  int f = t % F;
  int n = (t / F) % Nn;
  int b = t / (F * Nn);
  out[t] = in[((size_t)b * F + f) * Nn + n];     // (B,SEQ,N) -> (B,N,SEQ)
}

__global__ void k_zero(float* p, int n) {
  int t = blockIdx.x * blockDim.x + threadIdx.x;
  if (t < n) p[t] = 0.0f;
}

__global__ void k_deg(float* __restrict__ deg, const int* __restrict__ dst,
                      const float* __restrict__ w, int E) {
  int e = blockIdx.x * blockDim.x + threadIdx.x;
  if (e < E) atomicAdd(&deg[dst[e]], w[e]);
}

__global__ void k_dinv(const float* __restrict__ deg, float* __restrict__ dinv,
                       float* __restrict__ dinv2, int Nn) {
  int n = blockIdx.x * blockDim.x + threadIdx.x;
  if (n < Nn) {
    float di = rsqrtf(deg[n] + 1.0f);            // +1 self-loop weight
    dinv[n] = di;
    dinv2[n] = di * di;
  }
}

__global__ void k_norm(float* __restrict__ nrm, const float* __restrict__ dinv,
                       const int* __restrict__ src, const int* __restrict__ dst,
                       const float* __restrict__ w, int E) {
  int e = blockIdx.x * blockDim.x + threadIdx.x;
  if (e < E) nrm[e] = dinv[src[e]] * w[e] * dinv[dst[e]];
}

__global__ void k_gcn_init(float* __restrict__ out, const float* __restrict__ xw,
                           const float* __restrict__ dinv2, const float* __restrict__ bias,
                           int Nn, int Fo, int total) {
  int t = blockIdx.x * blockDim.x + threadIdx.x;
  if (t >= total) return;
  int f = t % Fo;
  int n = (t / Fo) % Nn;
  out[t] = xw[t] * dinv2[n] + bias[f];           // self-loop term + bias
}

__global__ void k_gcn_scatter(float* __restrict__ out, const float* __restrict__ xw,
                              const float* __restrict__ nrm,
                              const int* __restrict__ src, const int* __restrict__ dst,
                              int Nn, int E, int Fo, int total) {
  int t = blockIdx.x * blockDim.x + threadIdx.x;
  if (t >= total) return;
  int f = t % Fo;
  int e = (t / Fo) % E;
  int b = t / (Fo * E);
  int s = src[e], d = dst[e];
  atomicAdd(&out[((size_t)b * Nn + d) * Fo + f],
            xw[((size_t)b * Nn + s) * Fo + f] * nrm[e]);
}

__global__ void k_relu(float* p, int n) {
  int t = blockIdx.x * blockDim.x + threadIdx.x;
  if (t < n) p[t] = fmaxf(p[t], 0.0f);
}

__global__ void k_softmax(float* __restrict__ s, int cols, float scale) {
  __shared__ float red[128];
  float* p = s + (size_t)blockIdx.x * cols;
  int t = threadIdx.x;
  float v = p[t] * scale;
  red[t] = v; __syncthreads();
  for (int o = 64; o > 0; o >>= 1) { if (t < o) red[t] = fmaxf(red[t], red[t + o]); __syncthreads(); }
  float mx = red[0]; __syncthreads();
  float e = __expf(v - mx);
  red[t] = e; __syncthreads();
  for (int o = 64; o > 0; o >>= 1) { if (t < o) red[t] += red[t + o]; __syncthreads(); }
  p[t] = e * (1.0f / red[0]);
}

__global__ void k_layernorm(const float* __restrict__ x, const float* __restrict__ g,
                            const float* __restrict__ b, float* __restrict__ y, int cols) {
  __shared__ float rs[256], rq[256];
  int t = threadIdx.x;
  const float* p = x + (size_t)blockIdx.x * cols;
  float s = 0.0f, q = 0.0f;
  for (int i = t; i < cols; i += 256) { float v = p[i]; s += v; q += v * v; }
  rs[t] = s; rq[t] = q; __syncthreads();
  for (int o = 128; o > 0; o >>= 1) { if (t < o) { rs[t] += rs[t + o]; rq[t] += rq[t + o]; } __syncthreads(); }
  float mean = rs[0] / cols;
  float var  = rq[0] / cols - mean * mean;
  float rstd = rsqrtf(var + 1e-5f);
  for (int i = t; i < cols; i += 256) {
    float v = (p[i] - mean) * rstd;
    y[(size_t)blockIdx.x * cols + i] = v * g[i] + b[i];
  }
}

// ---------------------------------------------------------------------------
// Host orchestration
// ---------------------------------------------------------------------------
static inline unsigned cdiv(unsigned a, unsigned b) { return (a + b - 1) / b; }

extern "C" void kernel_launch(void* const* d_in, const int* in_sizes, int n_in,
                              void* d_out, int out_size, void* d_ws, size_t ws_size,
                              hipStream_t stream)
{
  (void)in_sizes; (void)n_in; (void)out_size; (void)ws_size;
  const int Bn = 128, SEQ = 96, NN = 150, E = 6000;
  const int D = 2400, S = 128, H = 3, DH = 800, DFF = 512, LABEL = 48, NL = 4;

  const float* inputs = (const float*)d_in[0];
  const int*   eidx   = (const int*)  d_in[1];
  const float* eattr  = (const float*)d_in[2];
  const int* srcp = eidx;
  const int* dstp = eidx + E;
  const float* gW[4]  = {(const float*)d_in[3], (const float*)d_in[5], (const float*)d_in[7], (const float*)d_in[9]};
  const float* gBs[4] = {(const float*)d_in[4], (const float*)d_in[6], (const float*)d_in[8], (const float*)d_in[10]};
  const float* Wq = (const float*)d_in[11], *Wk = (const float*)d_in[12];
  const float* Wv = (const float*)d_in[13], *Wo = (const float*)d_in[14];
  const float* bq = (const float*)d_in[15], *bk = (const float*)d_in[16];
  const float* bv = (const float*)d_in[17], *bo = (const float*)d_in[18];
  const float* ln1g = (const float*)d_in[19], *ln1b = (const float*)d_in[20];
  const float* ln2g = (const float*)d_in[21], *ln2b = (const float*)d_in[22];
  const float* fW1 = (const float*)d_in[23], *fb1 = (const float*)d_in[24];
  const float* fW2 = (const float*)d_in[25], *fb2 = (const float*)d_in[26];
  const float* outW = (const float*)d_in[27], *outb = (const float*)d_in[28];

  char* ws = (char*)d_ws;
  size_t off = 0;
  auto alloc = [&](size_t nfloats) -> float* {
    float* p = (float*)(ws + off);
    off += ((nfloats * 4 + 255) & ~(size_t)255);
    return p;
  };
  float* bufA  = alloc((size_t)Bn * NN * SEQ);   // x ping / final (B,2400)
  float* bufXW = alloc((size_t)Bn * NN * SEQ);   // gcn x@W
  float* bufC  = alloc((size_t)Bn * NN * SEQ);   // x pong
  float* q   = alloc((size_t)S * D);
  float* kb  = alloc((size_t)S * D);
  float* vb  = alloc((size_t)S * D);
  float* ob  = alloc((size_t)S * D);
  float* t1  = alloc((size_t)S * D);
  float* h1  = alloc((size_t)S * DFF);
  float* sc  = alloc((size_t)H * S * S);
  float* deg   = alloc(256);
  float* dinv  = alloc(256);
  float* dinv2 = alloc(256);
  float* nrm   = alloc(E);

  auto gemm = [&](const float* A, const float* B, const float* bias, const float* resid,
                  float* C, int M, int N, int K, int lda, int ldb, int ldc, int flags) {
    dim3 grid(cdiv(N, 64), cdiv(M, 16));
    gemm_bf16_wmma<<<grid, dim3(32), 0, stream>>>(A, B, bias, resid, C,
                                                  M, N, K, lda, ldb, ldc, flags);
  };

  // ---------- GCN preprocessing ----------
  {
    int tot = Bn * SEQ * NN;
    k_transpose_in<<<cdiv(tot, 256), 256, 0, stream>>>(inputs, bufA, Bn, SEQ, NN);
    k_zero<<<1, 256, 0, stream>>>(deg, NN);
    k_deg<<<cdiv(E, 256), 256, 0, stream>>>(deg, dstp, eattr, E);
    k_dinv<<<1, 256, 0, stream>>>(deg, dinv, dinv2, NN);
    k_norm<<<cdiv(E, 256), 256, 0, stream>>>(nrm, dinv, srcp, dstp, eattr, E);
  }

  // ---------- 4 GCN layers ----------
  const int dims[5] = {96, 72, 48, 24, 16};
  float* cur = bufA; float* other = bufC;
  for (int i = 0; i < 4; ++i) {
    int Fi = dims[i], Fo = dims[i + 1];
    gemm(cur, gW[i], nullptr, nullptr, bufXW, Bn * NN, Fo, Fi, Fi, Fo, Fo, 0);
    int tot = Bn * NN * Fo;
    k_gcn_init<<<cdiv(tot, 256), 256, 0, stream>>>(other, bufXW, dinv2, gBs[i], NN, Fo, tot);
    int tot2 = Bn * E * Fo;
    k_gcn_scatter<<<cdiv(tot2, 256), 256, 0, stream>>>(other, bufXW, nrm, srcp, dstp, NN, E, Fo, tot2);
    k_relu<<<cdiv(tot, 256), 256, 0, stream>>>(other, tot);
    float* tmp = cur; cur = other; other = tmp;
  }
  float* x = cur;   // == bufA : (128, 2400)

  // ---------- 4 transformer encoder layers ----------
  const float inv_sqrt_dh = 0.035355339059327376f;  // 1/sqrt(800)
  for (int l = 0; l < NL; ++l) {
    const float* wq = Wq + (size_t)l * D * D;  const float* bql = bq + (size_t)l * D;
    const float* wk = Wk + (size_t)l * D * D;  const float* bkl = bk + (size_t)l * D;
    const float* wv = Wv + (size_t)l * D * D;  const float* bvl = bv + (size_t)l * D;
    const float* wo = Wo + (size_t)l * D * D;  const float* bol = bo + (size_t)l * D;

    gemm(x, wq, bql, nullptr, q,  S, D, D, D, D, D, GF_BIAS);
    gemm(x, wk, bkl, nullptr, kb, S, D, D, D, D, D, GF_BIAS);
    gemm(x, wv, bvl, nullptr, vb, S, D, D, D, D, D, GF_BIAS);

    for (int h = 0; h < H; ++h)   // scores_h = Q_h @ K_h^T   (TRANSB)
      gemm(q + h * DH, kb + h * DH, nullptr, nullptr, sc + (size_t)h * S * S,
           S, S, DH, D, D, S, GF_TRANSB);
    k_softmax<<<H * S, 128, 0, stream>>>(sc, S, inv_sqrt_dh);
    for (int h = 0; h < H; ++h)   // O_h = attn_h @ V_h -> concat columns
      gemm(sc + (size_t)h * S * S, vb + h * DH, nullptr, nullptr, ob + h * DH,
           S, DH, S, S, D, D, 0);

    gemm(ob, wo, bol, x, t1, S, D, D, D, D, D, GF_BIAS | GF_RESID);
    k_layernorm<<<S, 256, 0, stream>>>(t1, ln1g + (size_t)l * D, ln1b + (size_t)l * D, x, D);

    gemm(x, fW1 + (size_t)l * D * DFF, fb1 + (size_t)l * DFF, nullptr, h1,
         S, DFF, D, D, DFF, DFF, GF_BIAS | GF_RELU);
    gemm(h1, fW2 + (size_t)l * DFF * D, fb2 + (size_t)l * D, x, t1,
         S, D, DFF, DFF, D, D, GF_BIAS | GF_RESID);
    k_layernorm<<<S, 256, 0, stream>>>(t1, ln2g + (size_t)l * D, ln2b + (size_t)l * D, x, D);
  }

  // ---------- output head ----------
  gemm(x, outW, outb, nullptr, (float*)d_out, S, LABEL, D, D, LABEL, LABEL, GF_BIAS);
}
